// VMambaDecoder_31275951849889
// MI455X (gfx1250) — matrix operand order repaired and checked by reading, not compiled, not run on hardware
//
#include <hip/hip_runtime.h>

// ---------------------------------------------------------------------------
// Types for CDNA5 WMMA (gfx1250, wave32)
// ---------------------------------------------------------------------------
typedef __attribute__((ext_vector_type(16))) __bf16 bf16x16;
typedef __attribute__((ext_vector_type(8)))  float  f32x8;

#define AC_NONE 0
#define AC_RELU 1
#define AC_GELU 2
#define AC_SIGM 3

__device__ __forceinline__ unsigned short f2bf(float f) {
  union { float f; unsigned int u; } x; x.f = f;
  return (unsigned short)((x.u + 0x7FFFu + ((x.u >> 16) & 1u)) >> 16);
}
__device__ __forceinline__ float sigm(float v) { return 1.f / (1.f + __expf(-v)); }

__device__ __forceinline__ float apply_act(float v, int act) {
  if (act == AC_RELU)      v = v > 0.f ? v : 0.f;
  else if (act == AC_GELU) { float t = tanhf(0.7978845608f * (v + 0.044715f * v * v * v));
                             v = 0.5f * v * (1.f + t); }
  else if (act == AC_SIGM) v = sigm(v);
  return v;
}

// Map scan-direction k and position l to a flat (h,w) row-major source index.
__device__ __forceinline__ int dirmap(int k, int l, int H, int W) {
  int L = H * W;
  int lp = (k >= 2) ? (L - 1 - l) : l;
  if (k & 1) { int wi = lp / H; int hi = lp - wi * H; return hi * W + wi; }
  return lp;
}

// LDS byte address (wave-relative) of a shared-memory pointer: flat addresses in
// the LDS aperture carry the LDS offset in the low 32 bits (ISA 10.2 aperture map).
__device__ __forceinline__ unsigned lds_addr_of(const void* p) {
  return (unsigned)(unsigned long long)p;
}

// Predicated load: clamp index so the load is always legal (v_cndmask on the
// index + unconditional load + v_cndmask on the data). Avoids per-element
// EXEC save/branch sequences around speculative-unsafe loads.
__device__ __forceinline__ unsigned short ld_masked(const unsigned short* base,
                                                    long long idx, bool ok) {
  unsigned short v = base[ok ? idx : 0];
  return ok ? v : (unsigned short)0;
}

// ---------------------------------------------------------------------------
// Shared WMMA micro-kernel: A in LDS as [co][k] (64x32, K pairs contiguous),
// B in LDS as [k][p] (32x64, row-major). Each wave: one 16-row A strip (mi),
// two 16-col B strips (nh) -> 2 WMMA.
// ---------------------------------------------------------------------------
__device__ __forceinline__ void wmma_tiles(const unsigned short* lds_a,
                                           const unsigned short* lds_b,
                                           int mi, int nh, int half, int lm,
                                           f32x8& acc0, f32x8& acc1) {
  union U { bf16x16 v; unsigned int d[8]; } Af, Bf0, Bf1;
  const unsigned int* la = (const unsigned int*)lds_a;
  int arow = mi * 16 + lm;
  int pc = nh * 32 + lm;
  #pragma unroll
  for (int v = 0; v < 8; ++v) {
    int kpA = ((v >> 2) << 3) + (half << 2) + (v & 3);   // A 16x32 dword pattern
    Af.d[v] = la[arow * 16 + kpA];
    int k = ((v >> 2) << 4) + (half << 3) + ((v & 3) << 1); // B 32x16 K-pair pattern
    unsigned lo0 = lds_b[k * 64 + pc],      hi0 = lds_b[(k + 1) * 64 + pc];
    unsigned lo1 = lds_b[k * 64 + pc + 16], hi1 = lds_b[(k + 1) * 64 + pc + 16];
    Bf0.d[v] = lo0 | (hi0 << 16);
    Bf1.d[v] = lo1 | (hi1 << 16);
  }
  acc0 = __builtin_amdgcn_wmma_f32_16x16x32_bf16(false, Af.v, false, Bf0.v,
                                                 (short)0, acc0, false, false);
  acc1 = __builtin_amdgcn_wmma_f32_16x16x32_bf16(false, Af.v, false, Bf1.v,
                                                 (short)0, acc1, false, false);
}

// ---------------------------------------------------------------------------
// Elementwise / conversion kernels
// ---------------------------------------------------------------------------
__global__ void k_f32_to_bf16(const float* __restrict__ in,
                              unsigned short* __restrict__ out, long long n) {
  long long i = (long long)blockIdx.x * blockDim.x + threadIdx.x;
  if (i < n) out[i] = f2bf(in[i]);
}

__global__ void k_add(const float* __restrict__ a, const float* __restrict__ b,
                      float* __restrict__ o, long long n) {
  long long i = (long long)blockIdx.x * blockDim.x + threadIdx.x;
  if (i < n) o[i] = a[i] + b[i];
}

__global__ void k_blend(const float* __restrict__ g, const float* __restrict__ o,
                        const float* __restrict__ h, float* __restrict__ out, long long n) {
  long long i = (long long)blockIdx.x * blockDim.x + threadIdx.x;
  if (i < n) { float gv = g[i]; out[i] = gv * o[i] + (1.f - gv) * h[i]; }
}

// dst[b, c0+c, p] = src[b, c, p]
__global__ void k_copy_ch(const float* __restrict__ s, float* __restrict__ d,
                          int Bn, int C, int HW, int dstC, int c0) {
  long long i = (long long)blockIdx.x * blockDim.x + threadIdx.x;
  long long n = (long long)Bn * C * HW;
  if (i >= n) return;
  int p = (int)(i % HW); long long t = i / HW; int c = (int)(t % C); int bb = (int)(t / C);
  d[((long long)bb * dstC + c0 + c) * HW + p] = s[i];
}

// ---------------------------------------------------------------------------
// LDS-staged bf16 WMMA GEMM for 1x1 convolutions.
//   Out(Co,Np) = W(Co,CI) * X(CI,Np), activations (B,CI,HW), HW % 64 == 0.
// ALIGNED (Co%64==0 && CI%32==0): stage via CDNA5 async global->LDS DMA
// (global_load_async_to_lds_b64, ASYNCcnt). Otherwise predicated scalar.
// ---------------------------------------------------------------------------
template <int CI, bool AL>
__global__ void k_gemm_wmma(const unsigned short* __restrict__ X,
                            const unsigned short* __restrict__ Wt,   // (Co, CI)
                            const float* __restrict__ bias,
                            float* __restrict__ Out,
                            int Bn, int HW, int Co, int act, float oscale) {
  __shared__ unsigned short lds_a[64 * 32];   // [co][k]
  __shared__ unsigned short lds_b[32 * 64];   // [k][p]
  const int tilesM = (Co + 63) >> 6;
  int mt = blockIdx.x % tilesM;
  long long nt = blockIdx.x / tilesM;
  int tid = threadIdx.x;
  int wv = tid >> 5, lane = tid & 31;
  int half = lane >> 4, lm = lane & 15;
  int mi = wv & 3, nh = wv >> 2;

  long long n0 = nt * 64;
  int bb = (int)(n0 / HW);
  int p0 = (int)(n0 - (long long)bb * HW);
  const unsigned short* Xb = X + (long long)bb * CI * HW + p0;

  f32x8 acc0 = {0.f,0.f,0.f,0.f,0.f,0.f,0.f,0.f};
  f32x8 acc1 = {0.f,0.f,0.f,0.f,0.f,0.f,0.f,0.f};

  #pragma unroll
  for (int k0 = 0; k0 < CI; k0 += 32) {
    if (AL) {
      // ---- async DMA staging: per thread 2x b64 for A, 2x b64 for B -------
      unsigned long long wb = (unsigned long long)Wt;
      unsigned long long xb = (unsigned long long)Xb;
      #pragma unroll
      for (int j = 0; j < 2; ++j) {
        int q = tid * 2 + j;                       // 0..511 qwords
        int co = q >> 3, kq = q & 7;               // A: 8 qwords per 32-k row
        unsigned aoff = (unsigned)((((long long)(mt * 64 + co)) * CI + k0 + kq * 4) * 2);
        unsigned adst = lds_addr_of(&lds_a[q * 4]);
        asm volatile("global_load_async_to_lds_b64 %0, %1, %2"
                     :: "v"(adst), "v"(aoff), "s"(wb) : "memory");
        int kb = q >> 4, po = (q & 15) * 4;        // B: 16 qwords per 64-p row
        unsigned boff = (unsigned)((((long long)(k0 + kb)) * HW + po) * 2);
        unsigned bdst = lds_addr_of(&lds_b[q * 4]);
        asm volatile("global_load_async_to_lds_b64 %0, %1, %2"
                     :: "v"(bdst), "v"(boff), "s"(xb) : "memory");
      }
      asm volatile("s_wait_asynccnt 0" ::: "memory");
      __syncthreads();
    } else {
      // ---- predicated scalar staging for ragged Co / CI -------------------
      #pragma unroll
      for (int i = 0; i < 8; ++i) {
        int idx = tid * 8 + i;                     // 0..2047
        int co = idx >> 5, kk = idx & 31;
        int gco = mt * 64 + co;
        bool okA = gco < Co && (k0 + kk) < CI;
        lds_a[idx] = ld_masked(Wt, (long long)gco * CI + k0 + kk, okA);
        int kb = idx >> 6, pp = idx & 63;
        bool okB = (k0 + kb) < CI;
        lds_b[idx] = ld_masked(Xb, (long long)(k0 + kb) * HW + pp, okB);
        if (k0 + 32 < CI && (k0 + 32 + kb) < CI)
          __builtin_prefetch(&Xb[(long long)(k0 + 32 + kb) * HW + pp], 0, 1);
      }
      __syncthreads();
    }
    wmma_tiles(lds_a, lds_b, mi, nh, half, lm, acc0, acc1);
    __syncthreads();
  }

  // ---- epilogue -----------------------------------------------------------
  int co_base = mt * 64 + mi * 16;
  long long outb = (long long)bb * Co * HW + p0;
  #pragma unroll
  for (int r = 0; r < 8; ++r) {
    int m = r + (half << 3);
    int co = co_base + m;
    if (co < Co) {
      float bv = bias ? bias[co] : 0.f;
      Out[outb + (long long)co * HW + nh * 32 + lm] =
          apply_act(acc0[r] + bv, act) * oscale;
      Out[outb + (long long)co * HW + nh * 32 + 16 + lm] =
          apply_act(acc1[r] + bv, act) * oscale;
    }
  }
}

// ---------------------------------------------------------------------------
// LDS-staged spatial conv (3x3 / 7x7, stride 1, "same" padding) with WMMA.
// Same 64(Co) x 64(pixel) block tile; per filter tap, stage a shifted
// zero-padded B tile into LDS (predicated loads) and accumulate.
// ---------------------------------------------------------------------------
template <int CI>
__global__ void k_conv_s(const unsigned short* __restrict__ X,   // (B,CI,H,W)
                         const unsigned short* __restrict__ Wt,  // (Co,CI,KH,KW)
                         const float* __restrict__ bias,
                         float* __restrict__ Out,                // (B,Co,H,W)
                         int Bn, int H, int W, int Co,
                         int KH, int KW, int pad, int act, float oscale) {
  __shared__ unsigned short lds_a[64 * 32];   // [co][k]
  __shared__ unsigned short lds_b[32 * 64];   // [k][p]
  const int HW = H * W;
  const int tilesM = (Co + 63) >> 6;
  int mt = blockIdx.x % tilesM;
  long long nt = blockIdx.x / tilesM;
  int tid = threadIdx.x;
  int wv = tid >> 5, lane = tid & 31;
  int half = lane >> 4, lm = lane & 15;
  int mi = wv & 3, nh = wv >> 2;

  long long n0 = nt * 64;
  int bb = (int)(n0 / HW);
  int p0 = (int)(n0 - (long long)bb * HW);
  const unsigned short* Xb = X + (long long)bb * CI * HW;

  f32x8 acc0 = {0.f,0.f,0.f,0.f,0.f,0.f,0.f,0.f};
  f32x8 acc1 = {0.f,0.f,0.f,0.f,0.f,0.f,0.f,0.f};

  for (int ky = 0; ky < KH; ++ky) {
    for (int kx = 0; kx < KW; ++kx) {
      #pragma unroll
      for (int k0 = 0; k0 < CI; k0 += 32) {
        #pragma unroll
        for (int i = 0; i < 8; ++i) {
          int idx = tid * 8 + i;
          int co = idx >> 5, kk = idx & 31;
          int gco = mt * 64 + co;
          bool okA = gco < Co && (k0 + kk) < CI;
          lds_a[idx] = ld_masked(
              Wt, ((((long long)gco) * CI + k0 + kk) * KH + ky) * KW + kx, okA);
          int kb = idx >> 6, pp = idx & 63;
          int pf = p0 + pp;
          int y = pf / W + ky - pad, x = pf % W + kx - pad;
          bool okB = (k0 + kb) < CI && y >= 0 && y < H && x >= 0 && x < W;
          lds_b[idx] = ld_masked(Xb, (long long)(k0 + kb) * HW + y * W + x, okB);
        }
        __syncthreads();
        wmma_tiles(lds_a, lds_b, mi, nh, half, lm, acc0, acc1);
        __syncthreads();
      }
    }
  }

  int co_base = mt * 64 + mi * 16;
  long long outb = (long long)bb * Co * HW + p0;
  #pragma unroll
  for (int r = 0; r < 8; ++r) {
    int m = r + (half << 3);
    int co = co_base + m;
    if (co < Co) {
      float bv = bias ? bias[co] : 0.f;
      Out[outb + (long long)co * HW + nh * 32 + lm] =
          apply_act(acc0[r] + bv, act) * oscale;
      Out[outb + (long long)co * HW + nh * 32 + 16 + lm] =
          apply_act(acc1[r] + bv, act) * oscale;
    }
  }
}

// ---------------------------------------------------------------------------
// Depthwise 3x3 conv + SiLU
// ---------------------------------------------------------------------------
__global__ void k_dw_silu(const float* __restrict__ x, const float* __restrict__ w,
                          const float* __restrict__ b, float* __restrict__ o,
                          int Bn, int C, int H, int W) {
  long long i = (long long)blockIdx.x * blockDim.x + threadIdx.x;
  long long n = (long long)Bn * C * H * W;
  if (i >= n) return;
  int px = (int)(i % W); long long t = i / W; int py = (int)(t % H); t /= H;
  int c = (int)(t % C); int bb = (int)(t / C);
  const float* xb = x + ((long long)bb * C + c) * H * W;
  float s = b[c];
  for (int ky = 0; ky < 3; ++ky)
    for (int kx = 0; kx < 3; ++kx) {
      int iy = py + ky - 1, ix = px + kx - 1;
      if (iy >= 0 && iy < H && ix >= 0 && ix < W) s += xb[iy * W + ix] * w[c * 9 + ky * 3 + kx];
    }
  o[i] = s * sigm(s);
}

// ---------------------------------------------------------------------------
// Channel-first LayerNorm (over C), one thread per (b, pixel). o may == x.
// ---------------------------------------------------------------------------
__global__ void k_ln(const float* __restrict__ x, float* __restrict__ o,
                     const float* __restrict__ g, const float* __restrict__ b,
                     int Bn, int C, int HW) {
  int i = blockIdx.x * blockDim.x + threadIdx.x;
  if (i >= Bn * HW) return;
  int bb = i / HW, p = i % HW;
  const float* xb = x + (long long)bb * C * HW + p;
  float m = 0.f;
  for (int c = 0; c < C; ++c) m += xb[(long long)c * HW];
  m /= C;
  float v = 0.f;
  for (int c = 0; c < C; ++c) { float d = xb[(long long)c * HW] - m; v += d * d; }
  v /= C;
  float rs = rsqrtf(v + 1e-5f);
  float* ob = o + (long long)bb * C * HW + p;
  for (int c = 0; c < C; ++c) ob[(long long)c * HW] = (xb[(long long)c * HW] - m) * rs * g[c] + b[c];
}

// ---------------------------------------------------------------------------
// pool2x: 3x3 window, stride 2, pad 1, mean (sum/9)
// ---------------------------------------------------------------------------
__global__ void k_pool2x(const float* __restrict__ x, float* __restrict__ o,
                         int Bn, int C, int H, int W) {
  int Ho = H / 2, Wo = W / 2;
  long long n = (long long)Bn * C * Ho * Wo;
  long long i = (long long)blockIdx.x * blockDim.x + threadIdx.x;
  if (i >= n) return;
  int px = (int)(i % Wo); long long t = i / Wo; int py = (int)(t % Ho); t /= Ho;
  int c = (int)(t % C); int bb = (int)(t / C);
  const float* xb = x + ((long long)bb * C + c) * H * W;
  float s = 0.f;
  for (int dy = -1; dy <= 1; ++dy)
    for (int dx = -1; dx <= 1; ++dx) {
      int iy = py * 2 + dy, ix = px * 2 + dx;
      if (iy >= 0 && iy < H && ix >= 0 && ix < W) s += xb[iy * W + ix];
    }
  o[i] = s / 9.f;
}

// ---------------------------------------------------------------------------
// Bilinear interpolation (linspace endpoints, as in reference)
// ---------------------------------------------------------------------------
__global__ void k_interp(const float* __restrict__ x, float* __restrict__ o,
                         int Bn, int C, int Hi, int Wi, int Ho, int Wo) {
  long long n = (long long)Bn * C * Ho * Wo;
  long long i = (long long)blockIdx.x * blockDim.x + threadIdx.x;
  if (i >= n) return;
  int px = (int)(i % Wo); long long t = i / Wo; int py = (int)(t % Ho); t /= Ho;
  int c = (int)(t % C); int bb = (int)(t / C);
  float yy = (Ho > 1) ? py * (float)(Hi - 1) / (float)(Ho - 1) : 0.f;
  float xx = (Wo > 1) ? px * (float)(Wi - 1) / (float)(Wo - 1) : 0.f;
  int y0 = (int)floorf(yy); int y1 = (y0 + 1 < Hi) ? y0 + 1 : Hi - 1;
  int x0 = (int)floorf(xx); int x1 = (x0 + 1 < Wi) ? x0 + 1 : Wi - 1;
  float wy = yy - y0, wx = xx - x0;
  const float* xb = x + ((long long)bb * C + c) * Hi * Wi;
  float a = xb[y0 * Wi + x0], bq = xb[y0 * Wi + x1];
  float cc = xb[y1 * Wi + x0], d  = xb[y1 * Wi + x1];
  o[i] = (a * (1.f - wx) + bq * wx) * (1.f - wy) + (cc * (1.f - wx) + d * wx) * wy;
}

// ---------------------------------------------------------------------------
// x_dbl[b,k,j,l] = sum_c xs[b,k,c,l] * xproj_w[k,j,c]   (C = 128, j in 0..9)
// ---------------------------------------------------------------------------
__global__ void k_xdbl(const float* __restrict__ xs, const float* __restrict__ xpw,
                       float* __restrict__ xd, int Bn, int H, int W) {
  int L = H * W;
  long long n = (long long)Bn * 4 * 10 * L;
  long long i = (long long)blockIdx.x * blockDim.x + threadIdx.x;
  if (i >= n) return;
  int l = (int)(i % L); long long t = i / L; int j = (int)(t % 10); t /= 10;
  int k = (int)(t % 4); int bb = (int)(t / 4);
  int src = dirmap(k, l, H, W);
  const float* xb = xs + (long long)bb * 128 * L + src;
  const float* wr = xpw + (k * 10 + j) * 128;
  float s = 0.f;
  for (int c = 0; c < 128; ++c) s += xb[(long long)c * L] * wr[c];
  xd[i] = s;
}

// ---------------------------------------------------------------------------
// Selective scan: one thread per (b,k,c) row, sequential over L.
// ---------------------------------------------------------------------------
__global__ void k_scan(const float* __restrict__ xd, const float* __restrict__ xs,
                       const float* __restrict__ dtw, const float* __restrict__ dtb,
                       const float* __restrict__ alog, const float* __restrict__ Dsp,
                       float* __restrict__ ys, int Bn, int H, int W) {
  int L = H * W;
  int i = blockIdx.x * blockDim.x + threadIdx.x;
  int n = Bn * 4 * 128;
  if (i >= n) return;
  int c = i % 128; int t = i / 128; int k = t % 4; int bb = t / 4;
  float wr[8];
  #pragma unroll
  for (int r = 0; r < 8; ++r) wr[r] = dtw[(k * 128 + c) * 8 + r];
  float db = dtb[k * 128 + c];
  float Aa = -__expf(alog[k * 128 + c]);
  float Dd = Dsp[k * 128 + c];
  const float* xdb = xd + (long long)(bb * 4 + k) * 10 * L;
  const float* xsb = xs + (long long)bb * 128 * L + (long long)c * L;
  float* yb = ys + ((long long)(bb * 4 + k) * 128 + c) * L;
  float hst = 0.f;
  for (int l = 0; l < L; ++l) {
    float dp = db;
    #pragma unroll
    for (int r = 0; r < 8; ++r) dp += xdb[(long long)r * L + l] * wr[r];
    float dt = (dp > 20.f) ? dp : log1pf(__expf(dp));      // softplus
    float Bv = xdb[8LL * L + l];
    float Cv = xdb[9LL * L + l];
    float xv = xsb[dirmap(k, l, H, W)];
    float a = __expf(dt * Aa);
    hst = a * hst + dt * Bv * xv;
    yb[l] = Cv * hst + Dd * xv;
  }
}

// ---------------------------------------------------------------------------
// Combine the four direction scans into y (b,128,H,W)
// ---------------------------------------------------------------------------
__global__ void k_combine(const float* __restrict__ ys, float* __restrict__ y,
                          int Bn, int H, int W) {
  int L = H * W;
  long long n = (long long)Bn * 128 * L;
  long long i = (long long)blockIdx.x * blockDim.x + threadIdx.x;
  if (i >= n) return;
  int l = (int)(i % L); long long t = i / L; int c = (int)(t % 128); int bb = (int)(t / 128);
  int hi = l / W, wi = l - hi * W;
  int l2 = wi * H + hi;
  const float* base = ys + (long long)bb * 4 * 128 * L;
  long long cL = (long long)c * L;
  float v = base[0LL * 128 * L + cL + l]
          + base[2LL * 128 * L + cL + (L - 1 - l)]
          + base[1LL * 128 * L + cL + l2]
          + base[3LL * 128 * L + cL + (L - 1 - l2)];
  y[i] = v;
}

// ---------------------------------------------------------------------------
// Host orchestration
// ---------------------------------------------------------------------------
enum {
  IN_NET0 = 0, IN_NET1, IN_NET2, IN_INP, IN_CORR, IN_FLOW,
  E_C1B, E_C1W, E_C2B, E_C2W, E_CFB, E_CFW, E_F1B, E_F1W, E_F2B, E_F2W,
  H_FL1B, H_FL1W, H_FL2B, H_FL2W, H_MK1B, H_MK1W, H_MK2B, H_MK2W,
  VM08_BASE = 24, VM16_BASE = 49, VM32_BASE = 74
};
enum {  // per-vmamba param offsets (pytree sorted keys)
  P_ALOG = 0, P_DS, P_DTB, P_DTW, P_DWB, P_DWW, P_GB, P_GW, P_INB, P_INW,
  P_IPB, P_IPW, P_M1B, P_M1W, P_M2B, P_M2W, P_N1B, P_N1G, P_N2B, P_N2G,
  P_ONB, P_ONG, P_OUTB, P_OUTW, P_XPW
};

struct VMW { unsigned short *ip, *inw, *gw, *m1, *m2, *outw; };

extern "C" void kernel_launch(void* const* d_in, const int* in_sizes, int n_in,
                              void* d_out_v, int out_size, void* d_ws, size_t ws_size,
                              hipStream_t stream) {
  (void)in_sizes; (void)n_in; (void)out_size; (void)ws_size;
  const int Bn = 4;
  const int H0 = 64, W0 = 128, H1 = 32, W1 = 64, H2 = 16, W2 = 32;
  const long long L0 = (long long)H0 * W0, L1 = (long long)H1 * W1, L2 = (long long)H2 * W2;

  auto F = [&](int i) { return (const float*)d_in[i]; };
  const float* net0 = F(IN_NET0);
  const float* net1 = F(IN_NET1);
  const float* net2 = F(IN_NET2);
  const float* corr = F(IN_CORR);
  const float* flow = F(IN_FLOW);

  float* o_net0  = (float*)d_out_v;                 // (4,128,64,128)
  float* o_net1  = o_net0 + 4194304;                // (4,128,32,64)
  float* o_net2  = o_net1 + 1048576;                // (4,128,16,32)
  float* o_mask  = o_net2 + 262144;                 // (4,144,64,128)
  float* o_dflow = o_mask + 4718592;                // (4,2,64,128)

  // --- workspace bump allocator ------------------------------------------
  size_t off = 0;
  auto alloc = [&](size_t bytes) -> void* {
    void* p = (char*)d_ws + off;
    off = (off + bytes + 255) & ~(size_t)255;
    return p;
  };
  auto grid1 = [&](long long n) { return dim3((unsigned)((n + 255) / 256)); };

  auto tobf = [&](const float* src, long long n, unsigned short* dst) {
    k_f32_to_bf16<<<grid1(n), 256, 0, stream>>>(src, dst, n);
  };
  auto cvtw = [&](const float* w, long long n) -> unsigned short* {
    unsigned short* p = (unsigned short*)alloc((size_t)n * 2);
    tobf(w, n, p);
    return p;
  };

  // --- convert all conv weights to bf16 ----------------------------------
  auto makeVMW = [&](int base, int ipCi) -> VMW {
    VMW w;
    w.ip   = cvtw(F(base + P_IPW),  128LL * ipCi);
    w.inw  = cvtw(F(base + P_INW),  128LL * 128);
    w.gw   = cvtw(F(base + P_GW),   128LL * 128);
    w.m1   = cvtw(F(base + P_M1W),  512LL * 128);
    w.m2   = cvtw(F(base + P_M2W),  128LL * 512);
    w.outw = cvtw(F(base + P_OUTW), 128LL * 128);
    return w;
  };
  VMW w32 = makeVMW(VM32_BASE, 256);
  VMW w16 = makeVMW(VM16_BASE, 384);
  VMW w08 = makeVMW(VM08_BASE, 384);
  unsigned short* w_c1  = cvtw(F(E_C1W),  64LL * 36);
  unsigned short* w_c2  = cvtw(F(E_C2W),  64LL * 64 * 9);
  unsigned short* w_f1  = cvtw(F(E_F1W),  64LL * 2 * 49);
  unsigned short* w_f2  = cvtw(F(E_F2W),  64LL * 64 * 9);
  unsigned short* w_cf  = cvtw(F(E_CFW),  126LL * 128 * 9);
  unsigned short* w_fl1 = cvtw(F(H_FL1W), 256LL * 128 * 9);
  unsigned short* w_fl2 = cvtw(F(H_FL2W), 2LL * 256 * 9);
  unsigned short* w_mk1 = cvtw(F(H_MK1W), 256LL * 128 * 9);
  unsigned short* w_mk2 = cvtw(F(H_MK2W), 144LL * 256);

  // --- activation buffers (sized for the largest scale, reused) ----------
  float* BufA = (float*)alloc((size_t)Bn * 384 * L0 * 4);    // concat input
  float* BufB = (float*)alloc((size_t)Bn * 512 * L0 * 4);    // ys / m1 / 256-ch heads
  float* F1 = (float*)alloc((size_t)Bn * 128 * L0 * 4);
  float* F2 = (float*)alloc((size_t)Bn * 128 * L0 * 4);
  float* F3 = (float*)alloc((size_t)Bn * 128 * L0 * 4);
  float* F4 = (float*)alloc((size_t)Bn * 128 * L0 * 4);
  float* G1 = (float*)alloc((size_t)Bn * 128 * L0 * 4);      // mf (persists into vm08)
  float* G2 = (float*)alloc((size_t)Bn * 128 * L0 * 4);      // interp buffer
  float* G3 = (float*)alloc((size_t)Bn * 128 * L0 * 4);      // pool buffer
  float* D40 = (float*)alloc((size_t)Bn * 40 * L0 * 4);      // x_dbl
  unsigned short* BFA = (unsigned short*)alloc((size_t)Bn * 512 * L0 * 2);
  unsigned short* BF2 = (unsigned short*)alloc((size_t)Bn * 128 * L0 * 2);

  // --- launch helpers -----------------------------------------------------
  auto gemm = [&](const unsigned short* Xbf, const unsigned short* Wbf, const float* bias,
                  float* out, int Ci, int HW, int Co, int act, float sc) {
    long long Np = (long long)Bn * HW;             // HW % 64 == 0 for all scales
    int tilesM = (Co + 63) / 64;
    dim3 g((unsigned)(tilesM * (Np / 64)));
    switch (Ci) {
      case 36:  k_gemm_wmma<36, false><<<g, 256, 0, stream>>>(Xbf, Wbf, bias, out, Bn, HW, Co, act, sc); break;
      case 128: k_gemm_wmma<128, true><<<g, 256, 0, stream>>>(Xbf, Wbf, bias, out, Bn, HW, Co, act, sc); break;
      case 256:
        if (Co % 64 == 0) k_gemm_wmma<256, true><<<g, 256, 0, stream>>>(Xbf, Wbf, bias, out, Bn, HW, Co, act, sc);
        else              k_gemm_wmma<256, false><<<g, 256, 0, stream>>>(Xbf, Wbf, bias, out, Bn, HW, Co, act, sc);
        break;
      case 384: k_gemm_wmma<384, true><<<g, 256, 0, stream>>>(Xbf, Wbf, bias, out, Bn, HW, Co, act, sc); break;
      case 512: k_gemm_wmma<512, true><<<g, 256, 0, stream>>>(Xbf, Wbf, bias, out, Bn, HW, Co, act, sc); break;
      default:  break;
    }
  };
  auto convs = [&](const unsigned short* Xbf, const unsigned short* Wbf, const float* bias,
                   float* out, int Ci, int Hh, int Ww, int Co, int KH, int KW, int pad,
                   int act, float sc) {
    long long Np = (long long)Bn * Hh * Ww;
    int tilesM = (Co + 63) / 64;
    dim3 g((unsigned)(tilesM * (Np / 64)));
    switch (Ci) {
      case 2:   k_conv_s<2><<<g, 256, 0, stream>>>(Xbf, Wbf, bias, out, Bn, Hh, Ww, Co, KH, KW, pad, act, sc); break;
      case 64:  k_conv_s<64><<<g, 256, 0, stream>>>(Xbf, Wbf, bias, out, Bn, Hh, Ww, Co, KH, KW, pad, act, sc); break;
      case 128: k_conv_s<128><<<g, 256, 0, stream>>>(Xbf, Wbf, bias, out, Bn, Hh, Ww, Co, KH, KW, pad, act, sc); break;
      case 256: k_conv_s<256><<<g, 256, 0, stream>>>(Xbf, Wbf, bias, out, Bn, Hh, Ww, Co, KH, KW, pad, act, sc); break;
      default:  break;
    }
  };
  auto copych = [&](const float* s, float* d, int C, int HW, int dstC, int c0) {
    k_copy_ch<<<grid1((long long)Bn * C * HW), 256, 0, stream>>>(s, d, Bn, C, HW, dstC, c0);
  };
  auto ln = [&](const float* x, float* o, const float* g, const float* b, int HW) {
    k_ln<<<grid1((long long)Bn * HW), 256, 0, stream>>>(x, o, g, b, Bn, 128, HW);
  };
  auto addk = [&](const float* a, const float* b, float* o, long long n) {
    k_add<<<grid1(n), 256, 0, stream>>>(a, b, o, n);
  };

  // --- VSS block (SS2D + MLP), xio (=feat) updated in place to the output -
  auto run_vss = [&](int base, const VMW& w, float* xio, int Hh, int Ww) {
    int HW = Hh * Ww;
    long long n128 = (long long)Bn * 128 * HW;
    ln(xio, F2, F(base + P_N1G), F(base + P_N1B), HW);                                // t1
    tobf(F2, n128, BFA);
    gemm(BFA, w.inw, F(base + P_INB), F3, 128, HW, 128, AC_NONE, 1.f);                // u0
    k_dw_silu<<<grid1(n128), 256, 0, stream>>>(F3, F(base + P_DWW), F(base + P_DWB),
                                               F2, Bn, 128, Hh, Ww);                  // xsilu->F2
    k_xdbl<<<grid1((long long)Bn * 40 * HW), 256, 0, stream>>>(F2, F(base + P_XPW), D40, Bn, Hh, Ww);
    k_scan<<<grid1(Bn * 4 * 128), 256, 0, stream>>>(D40, F2, F(base + P_DTW), F(base + P_DTB),
                                                    F(base + P_ALOG), F(base + P_DS),
                                                    BufB, Bn, Hh, Ww);                // ys->BufB
    k_combine<<<grid1(n128), 256, 0, stream>>>(BufB, F3, Bn, Hh, Ww);                 // y->F3
    ln(F3, F3, F(base + P_ONG), F(base + P_ONB), HW);                                 // in place
    tobf(F3, n128, BFA);
    gemm(BFA, w.outw, F(base + P_OUTB), F4, 128, HW, 128, AC_NONE, 1.f);              // s
    addk(xio, F4, xio, n128);                                                         // x2
    ln(xio, F4, F(base + P_N2G), F(base + P_N2B), HW);                                // t2
    tobf(F4, n128, BFA);
    gemm(BFA, w.m1, F(base + P_M1B), BufB, 128, HW, 512, AC_GELU, 1.f);               // gelu(m1)
    tobf(BufB, (long long)Bn * 512 * HW, BFA);
    gemm(BFA, w.m2, F(base + P_M2B), F2, 512, HW, 128, AC_NONE, 1.f);
    addk(xio, F2, xio, n128);                                                         // x3
  };

  // --- conv_vmamba: concat -> ip conv -> vss -> sigmoid gate blend --------
  auto run_cvm = [&](int base, const VMW& w, const float* h, const float* x1,
                     const float* x2, int in_ch, int Hh, int Ww, float* result) {
    int HW = Hh * Ww;
    long long n128 = (long long)Bn * 128 * HW;
    copych(h, BufA, 128, HW, in_ch, 0);
    copych(x1, BufA, 128, HW, in_ch, 128);
    if (x2) copych(x2, BufA, 128, HW, in_ch, 256);
    tobf(BufA, (long long)Bn * in_ch * HW, BFA);
    gemm(BFA, w.ip, F(base + P_IPB), F1, in_ch, HW, 128, AC_NONE, 1.f);               // feat
    run_vss(base, w, F1, Hh, Ww);                                                     // out in F1
    tobf(F1, n128, BFA);
    gemm(BFA, w.gw, F(base + P_GB), F2, 128, HW, 128, AC_SIGM, 1.f);                  // g
    k_blend<<<grid1(n128), 256, 0, stream>>>(F2, F1, h, result, n128);
  };

  // ===================== forward =========================================
  // Stage A: net2' = conv_vmamba(vm32, net2, pool2x(net1))
  k_pool2x<<<grid1((long long)Bn * 128 * L2), 256, 0, stream>>>(net1, G2, Bn, 128, H1, W1);
  run_cvm(VM32_BASE, w32, net2, G2, nullptr, 256, H2, W2, o_net2);

  // Stage B: net1' = conv_vmamba(vm16, net1, pool2x(net0), interp(net2'))
  k_pool2x<<<grid1((long long)Bn * 128 * L1), 256, 0, stream>>>(net0, G3, Bn, 128, H0, W0);
  k_interp<<<grid1((long long)Bn * 128 * L1), 256, 0, stream>>>(o_net2, G2, Bn, 128, H2, W2, H1, W1);
  run_cvm(VM16_BASE, w16, net1, G3, G2, 384, H1, W1, o_net1);

  // Stage C: motion encoder -> mf (G1, 128 channels)
  {
    long long n64 = (long long)Bn * 64 * L0;
    tobf(corr, (long long)Bn * 36 * L0, BFA);
    gemm(BFA, w_c1, F(E_C1B), F1, 36, (int)L0, 64, AC_RELU, 1.f);
    tobf(F1, n64, BFA);
    convs(BFA, w_c2, F(E_C2B), F2, 64, H0, W0, 64, 3, 3, 1, AC_RELU, 1.f);   // c -> F2
    tobf(flow, (long long)Bn * 2 * L0, BFA);
    convs(BFA, w_f1, F(E_F1B), F1, 2, H0, W0, 64, 7, 7, 3, AC_RELU, 1.f);
    tobf(F1, n64, BFA);
    convs(BFA, w_f2, F(E_F2B), F3, 64, H0, W0, 64, 3, 3, 1, AC_RELU, 1.f);   // f -> F3
    copych(F2, F4, 64, (int)L0, 128, 0);
    copych(F3, F4, 64, (int)L0, 128, 64);                                    // concat(c,f)
    tobf(F4, (long long)Bn * 128 * L0, BFA);
    convs(BFA, w_cf, F(E_CFB), F1, 128, H0, W0, 126, 3, 3, 1, AC_RELU, 1.f); // m (126ch)
    copych(F1, G1, 126, (int)L0, 128, 0);
    copych(flow, G1, 2, (int)L0, 128, 126);                                  // mf = [m, flow]
  }

  // Stage D: net0' = conv_vmamba(vm08, net0, mf, interp(net1'))
  k_interp<<<grid1((long long)Bn * 128 * L0), 256, 0, stream>>>(o_net1, G2, Bn, 128, H1, W1, H0, W0);
  run_cvm(VM08_BASE, w08, net0, G1, G2, 384, H0, W0, o_net0);

  // Stage E: heads
  tobf(o_net0, (long long)Bn * 128 * L0, BF2);
  convs(BF2, w_fl1, F(H_FL1B), BufB, 128, H0, W0, 256, 3, 3, 1, AC_RELU, 1.f);
  tobf(BufB, (long long)Bn * 256 * L0, BFA);
  convs(BFA, w_fl2, F(H_FL2B), o_dflow, 256, H0, W0, 2, 3, 3, 1, AC_NONE, 1.f);
  convs(BF2, w_mk1, F(H_MK1B), BufB, 128, H0, W0, 256, 3, 3, 1, AC_RELU, 1.f);
  tobf(BufB, (long long)Bn * 256 * L0, BFA);
  convs(BFA, w_mk2, F(H_MK2B), o_mask, 256, H0, W0, 144, 1, 1, 0, AC_NONE, 0.25f);
}
